// RNN_19894288515603
// MI455X (gfx1250) — matrix operand order repaired and checked by reading, not compiled, not run on hardware
//
#include <hip/hip_runtime.h>

#define SEQ 512
#define BATCH 256
#define IN_DIM 300
#define HID 128
#define ALPHA 0.1f

typedef __attribute__((ext_vector_type(16))) __bf16 v16bf;
typedef __attribute__((ext_vector_type(8)))  float  v8f;
typedef __attribute__((ext_vector_type(4)))  float  f32x4;
typedef __attribute__((ext_vector_type(8)))  unsigned short u16x8;

union V16BF {
  v16bf v;
  u16x8 h[2];
  __bf16 b[16];
  unsigned short u[16];
};

__device__ __forceinline__ void split_bf16(float x, __bf16& hi, __bf16& lo) {
  hi = (__bf16)x;
  lo = (__bf16)(x - (float)hi);
}
__device__ __forceinline__ unsigned short bfbits(__bf16 v) {
  return __builtin_bit_cast(unsigned short, v);
}
__device__ __forceinline__ v8f wmma_bf16(v16bf a, v16bf b, v8f c) {
  return __builtin_amdgcn_wmma_f32_16x16x32_bf16(
      /*neg_a=*/false, a, /*neg_b=*/false, b,
      /*c_mod=*/(short)0, c, /*reuse_a=*/false, /*reuse_b=*/false);
}
__device__ __forceinline__ float leaky(float x) { return x > 0.f ? x : ALPHA * x; }

// ---------------------------------------------------------------------------
// Kernel A: xp[m][n] = sum_k x[m][k] * W_ih[n][k] + b_ih[n] + b_hh[n]
// M = SEQ*BATCH = 131072, K = 300 (padded to 320 = 10 chunks of 32), N = 128.
// Each WG (256 thr, 8 waves) does a 128-row M-tile; wave w does rows w*16..+15
// for all 8 N-tiles. W_ih chunks are staged to LDS as split-bf16, double-buffered.
// ---------------------------------------------------------------------------
__global__ __launch_bounds__(256) void proj_kernel(
    const float* __restrict__ x, const float* __restrict__ W_ih,
    const float* __restrict__ b_ih, const float* __restrict__ b_hh,
    float* __restrict__ xp)
{
  __shared__ unsigned short sBhi[2][HID * 32];
  __shared__ unsigned short sBlo[2][HID * 32];

  const int tid  = threadIdx.x;
  const int lane = tid & 31;
  const int wave = tid >> 5;
  const int l15  = lane & 15;
  const int lh   = lane >> 4;

  auto stage = [&](int c, int buf) {
    for (int idx = tid; idx < HID * 32; idx += 256) {
      int n  = idx >> 5;
      int kk = idx & 31;
      int k  = 32 * c + kk;
      float w = (k < IN_DIM) ? W_ih[n * IN_DIM + k] : 0.f;
      __bf16 hi, lo; split_bf16(w, hi, lo);
      sBhi[buf][idx] = bfbits(hi);
      sBlo[buf][idx] = bfbits(lo);
    }
  };

  stage(0, 0);

  const long m0 = (long)blockIdx.x * 128 + wave * 16;
  const float* xrow = x + (m0 + l15) * (long)IN_DIM;

  v8f acc[8];
  #pragma unroll
  for (int nt = 0; nt < 8; ++nt)
    #pragma unroll
    for (int j = 0; j < 8; ++j) acc[nt][j] = 0.f;

  __syncthreads();

  for (int c = 0; c < 10; ++c) {
    if (c + 1 < 10) stage(c + 1, (c + 1) & 1);

    // A-matrix: lane row = l15, K segments [k1..k1+7] and [k1+16..k1+23]
    const int k1 = 32 * c + 8 * lh;
    float a1[8], a2[8];
    if (c < 9) {
      f32x4 p0 = *(const f32x4*)(xrow + k1);
      f32x4 p1 = *(const f32x4*)(xrow + k1 + 4);
      f32x4 q0 = *(const f32x4*)(xrow + k1 + 16);
      f32x4 q1 = *(const f32x4*)(xrow + k1 + 20);
      #pragma unroll
      for (int i = 0; i < 4; ++i) {
        a1[i] = p0[i]; a1[4 + i] = p1[i];
        a2[i] = q0[i]; a2[4 + i] = q1[i];
      }
    } else {  // tail chunk: guard K >= 300
      #pragma unroll
      for (int i = 0; i < 8; ++i) {
        int ka = k1 + i;
        int kb = k1 + 16 + i;
        a1[i] = (ka < IN_DIM) ? xrow[ka] : 0.f;
        a2[i] = (kb < IN_DIM) ? xrow[kb] : 0.f;
      }
    }

    V16BF Ahi, Alo;
    #pragma unroll
    for (int i = 0; i < 8; ++i) {
      __bf16 hi, lo;
      split_bf16(a1[i], hi, lo); Ahi.b[i] = hi;     Alo.b[i] = lo;
      split_bf16(a2[i], hi, lo); Ahi.b[8 + i] = hi; Alo.b[8 + i] = lo;
    }

    const int buf  = c & 1;
    const int koff = 8 * lh;  // B: lanes 0-15 -> K 0..7,16..23; lanes 16-31 -> K 8..15,24..31
    #pragma unroll
    for (int nt = 0; nt < 8; ++nt) {
      const int col = nt * 16 + l15;
      V16BF Bhi, Blo;
      const u16x8* ph = (const u16x8*)&sBhi[buf][col * 32 + koff];
      const u16x8* pl = (const u16x8*)&sBlo[buf][col * 32 + koff];
      Bhi.h[0] = ph[0]; Bhi.h[1] = ph[2];
      Blo.h[0] = pl[0]; Blo.h[1] = pl[2];
      acc[nt] = wmma_bf16(Ahi.v, Bhi.v, acc[nt]);
      acc[nt] = wmma_bf16(Ahi.v, Blo.v, acc[nt]);
      acc[nt] = wmma_bf16(Alo.v, Bhi.v, acc[nt]);
    }
    __syncthreads();
  }

  // C layout: vgpr j, lanes 0-15: M=j, N=lane; lanes 16-31: M=8+j, N=lane-16
  #pragma unroll
  for (int nt = 0; nt < 8; ++nt) {
    const int col = nt * 16 + l15;
    const float bias = b_ih[col] + b_hh[col];
    #pragma unroll
    for (int j = 0; j < 8; ++j) {
      const long m = m0 + j + 8 * lh;
      xp[m * HID + col] = acc[nt][j] + bias;
    }
  }
}

// ---------------------------------------------------------------------------
// Kernel B: sequential scan, h = tanh(xp[t] + h @ W_hh^T), hs[t] overwrites
// xp[t] in place.  16 WGs (one per 16-batch tile) x 8 waves (one 16-col N-tile
// each).  W_hh slice resident in VGPRs; h exchanged via double-buffered LDS
// (split bf16); xp[t+1] prefetched into registers during step t's WMMA chain.
// Three independent accumulators (one per split-bf16 product term) cut the
// per-step dependent WMMA chain from 12 to 4.
// ---------------------------------------------------------------------------
__global__ __launch_bounds__(256) void scan_kernel(
    const float* __restrict__ W_hh, float* xp)
{
  __shared__ unsigned short hbuf[2][2][16][HID];  // [buf][hi/lo][m][k]

  const int tid = threadIdx.x, lane = tid & 31, wave = tid >> 5;
  const int l15 = lane & 15, lh = lane >> 4;
  const int b0 = blockIdx.x * 16;
  const int colg = wave * 16 + l15;

  // Resident B = W_hh chunks: B[k][n] = W_hh[n][k]; lane col n = colg,
  // 16 contiguous K starting at 32*c + 16*lh.
  V16BF Bhi[4], Blo[4];
  #pragma unroll
  for (int c = 0; c < 4; ++c) {
    const float* wp = W_hh + colg * HID + 32 * c + 16 * lh;
    #pragma unroll
    for (int i = 0; i < 16; ++i) {
      __bf16 hi, lo; split_bf16(wp[i], hi, lo);
      Bhi[c].b[i] = hi; Blo[c].b[i] = lo;
    }
  }

  for (int idx = tid; idx < 2 * 2 * 16 * HID; idx += 256)
    ((unsigned short*)hbuf)[idx] = 0;  // h0 = 0
  __syncthreads();

  float xn[8];
  {
    const float* src = xp + ((long)b0 + 8 * lh) * HID + colg;
    #pragma unroll
    for (int j = 0; j < 8; ++j) xn[j] = src[j * HID];
  }

  for (int t = 0; t < SEQ; ++t) {
    v8f accHH, accHL, accLH;
    #pragma unroll
    for (int j = 0; j < 8; ++j) {
      accHH[j] = xn[j];   // fold xp[t] into the hi*hi chain
      accHL[j] = 0.f;
      accLH[j] = 0.f;
    }

    if (t + 1 < SEQ) {  // prefetch next xp tile, overlaps WMMA chains
      const float* src = xp + ((long)(t + 1) * BATCH + b0 + 8 * lh) * HID + colg;
      #pragma unroll
      for (int j = 0; j < 8; ++j) xn[j] = src[j * HID];
    }

    const int rb = (t + 1) & 1;  // == (t-1)&1
    #pragma unroll
    for (int c = 0; c < 4; ++c) {
      const int kb = 32 * c + 16 * lh;
      V16BF Ahi, Alo;
      const u16x8* ph = (const u16x8*)&hbuf[rb][0][l15][kb];
      const u16x8* pl = (const u16x8*)&hbuf[rb][1][l15][kb];
      Ahi.h[0] = ph[0]; Ahi.h[1] = ph[1];  // K kb..kb+15
      Alo.h[0] = pl[0]; Alo.h[1] = pl[1];
      accHH = wmma_bf16(Ahi.v, Bhi[c].v, accHH);   // three independent
      accHL = wmma_bf16(Ahi.v, Blo[c].v, accHL);   // 4-deep chains
      accLH = wmma_bf16(Alo.v, Bhi[c].v, accLH);
    }

    const int wb = t & 1;
    float* dst = xp + ((long)t * BATCH + b0 + 8 * lh) * HID + colg;  // hs[t]
    #pragma unroll
    for (int j = 0; j < 8; ++j) {
      float h = tanhf(accHH[j] + (accHL[j] + accLH[j]));
      __bf16 hi, lo; split_bf16(h, hi, lo);
      hbuf[wb][0][j + 8 * lh][colg] = bfbits(hi);
      hbuf[wb][1][j + 8 * lh][colg] = bfbits(lo);
      dst[j * HID] = h;
    }
    __syncthreads();
  }
}

// ---------------------------------------------------------------------------
// Kernel C: MLP head. Layer 1 (128->64) via split-bf16 WMMA per 16-row tile,
// layers 2..5 scalar per lane (row) through LDS. 4 waves/WG, 4 tiles/WG.
// ---------------------------------------------------------------------------
__global__ __launch_bounds__(128) void mlp_kernel(
    const float* __restrict__ hs,
    const float* __restrict__ W1, const float* __restrict__ b1,
    const float* __restrict__ W2, const float* __restrict__ b2,
    const float* __restrict__ W3, const float* __restrict__ b3,
    const float* __restrict__ W4, const float* __restrict__ b4,
    const float* __restrict__ W5, const float* __restrict__ b5,
    float* __restrict__ out)
{
  __shared__ unsigned short sW1hi[64 * HID];
  __shared__ unsigned short sW1lo[64 * HID];
  __shared__ float sW2[32 * 64];
  __shared__ float sW3[16 * 32];
  __shared__ float sW4[8 * 16];
  __shared__ float sW5[8];
  __shared__ float sb1[64], sb2[32], sb3[16], sb4[8], sb5[1];
  __shared__ float y1[4][16][64];

  const int tid = threadIdx.x, lane = tid & 31, wave = tid >> 5;
  const int l15 = lane & 15, lh = lane >> 4;

  for (int idx = tid; idx < 64 * HID; idx += 128) {
    __bf16 hi, lo; split_bf16(W1[idx], hi, lo);
    sW1hi[idx] = bfbits(hi);
    sW1lo[idx] = bfbits(lo);
  }
  for (int idx = tid; idx < 32 * 64; idx += 128) sW2[idx] = W2[idx];
  for (int idx = tid; idx < 16 * 32; idx += 128) sW3[idx] = W3[idx];
  if (tid < 8 * 16) sW4[tid] = W4[tid];
  if (tid < 8)  sW5[tid] = W5[tid];
  if (tid < 64) sb1[tid] = b1[tid];
  if (tid < 32) sb2[tid] = b2[tid];
  if (tid < 16) sb3[tid] = b3[tid];
  if (tid < 8)  sb4[tid] = b4[tid];
  if (tid == 0) sb5[0] = b5[0];
  __syncthreads();

  const long tile = (long)blockIdx.x * 4 + wave;
  const long m0 = tile * 16;
  const float* arow = hs + (m0 + l15) * HID;

  v8f acc[4];
  #pragma unroll
  for (int nt = 0; nt < 4; ++nt)
    #pragma unroll
    for (int j = 0; j < 8; ++j) acc[nt][j] = 0.f;

  #pragma unroll
  for (int c = 0; c < 4; ++c) {
    const int k1 = 32 * c + 8 * lh;
    f32x4 p0 = *(const f32x4*)(arow + k1);
    f32x4 p1 = *(const f32x4*)(arow + k1 + 4);
    f32x4 q0 = *(const f32x4*)(arow + k1 + 16);
    f32x4 q1 = *(const f32x4*)(arow + k1 + 20);
    V16BF Ahi, Alo;
    #pragma unroll
    for (int i = 0; i < 4; ++i) {
      __bf16 hi, lo;
      split_bf16(p0[i], hi, lo); Ahi.b[i] = hi;      Alo.b[i] = lo;
      split_bf16(p1[i], hi, lo); Ahi.b[4 + i] = hi;  Alo.b[4 + i] = lo;
      split_bf16(q0[i], hi, lo); Ahi.b[8 + i] = hi;  Alo.b[8 + i] = lo;
      split_bf16(q1[i], hi, lo); Ahi.b[12 + i] = hi; Alo.b[12 + i] = lo;
    }
    const int koff = 32 * c + 8 * lh;
    #pragma unroll
    for (int nt = 0; nt < 4; ++nt) {
      const int col = nt * 16 + l15;
      V16BF Bhi, Blo;
      const u16x8* ph = (const u16x8*)&sW1hi[col * HID + koff];
      const u16x8* pl = (const u16x8*)&sW1lo[col * HID + koff];
      Bhi.h[0] = ph[0]; Bhi.h[1] = ph[2];
      Blo.h[0] = pl[0]; Blo.h[1] = pl[2];
      acc[nt] = wmma_bf16(Ahi.v, Bhi.v, acc[nt]);
      acc[nt] = wmma_bf16(Ahi.v, Blo.v, acc[nt]);
      acc[nt] = wmma_bf16(Alo.v, Bhi.v, acc[nt]);
    }
  }

  #pragma unroll
  for (int nt = 0; nt < 4; ++nt) {
    const int col = nt * 16 + l15;
    const float bb = sb1[col];
    #pragma unroll
    for (int j = 0; j < 8; ++j)
      y1[wave][j + 8 * lh][col] = leaky(acc[nt][j] + bb);
  }
  __syncthreads();

  if (lane < 16) {
    const float* yr = y1[wave][lane];
    float v2[32];
    #pragma unroll
    for (int o = 0; o < 32; ++o) {
      float s = sb2[o];
      for (int k = 0; k < 64; ++k) s = fmaf(sW2[o * 64 + k], yr[k], s);
      v2[o] = leaky(s);
    }
    float v3[16];
    #pragma unroll
    for (int o = 0; o < 16; ++o) {
      float s = sb3[o];
      #pragma unroll
      for (int k = 0; k < 32; ++k) s = fmaf(sW3[o * 32 + k], v2[k], s);
      v3[o] = leaky(s);
    }
    float v4[8];
    #pragma unroll
    for (int o = 0; o < 8; ++o) {
      float s = sb4[o];
      #pragma unroll
      for (int k = 0; k < 16; ++k) s = fmaf(sW4[o * 16 + k], v3[k], s);
      v4[o] = leaky(s);
    }
    float s = sb5[0];
    #pragma unroll
    for (int k = 0; k < 8; ++k) s = fmaf(sW5[k], v4[k], s);
    out[m0 + lane] = leaky(s);
  }
}

extern "C" void kernel_launch(void* const* d_in, const int* in_sizes, int n_in,
                              void* d_out, int out_size, void* d_ws, size_t ws_size,
                              hipStream_t stream) {
  (void)in_sizes; (void)n_in; (void)out_size; (void)ws_size;
  const float* x    = (const float*)d_in[0];
  const float* W_ih = (const float*)d_in[1];
  const float* W_hh = (const float*)d_in[2];
  const float* b_ih = (const float*)d_in[3];
  const float* b_hh = (const float*)d_in[4];
  const float* W1 = (const float*)d_in[5];  const float* b1 = (const float*)d_in[6];
  const float* W2 = (const float*)d_in[7];  const float* b2 = (const float*)d_in[8];
  const float* W3 = (const float*)d_in[9];  const float* b3 = (const float*)d_in[10];
  const float* W4 = (const float*)d_in[11]; const float* b4 = (const float*)d_in[12];
  const float* W5 = (const float*)d_in[13]; const float* b5 = (const float*)d_in[14];

  // workspace: xp = (SEQ*BATCH, HID) f32 = 64 MiB; scan overwrites it with hs.
  float* xp = (float*)d_ws;

  proj_kernel<<<dim3((SEQ * BATCH) / 128), dim3(256), 0, stream>>>(
      x, W_ih, b_ih, b_hh, xp);
  scan_kernel<<<dim3(BATCH / 16), dim3(256), 0, stream>>>(W_hh, xp);
  mlp_kernel<<<dim3((SEQ * BATCH) / 64), dim3(128), 0, stream>>>(
      xp, W1, b1, W2, b2, W3, b3, W4, b4, W5, b5, (float*)d_out);
}
